// AV_VQVAE_Encoder_43044162240520
// MI455X (gfx1250) — compile-verified, matrix operand order checked by reading, not run on hardware
//
#include <hip/hip_runtime.h>
#include <stdint.h>

// ---------- types ----------
typedef __attribute__((ext_vector_type(16))) __bf16 bf16x16;
typedef __attribute__((ext_vector_type(8)))  float  f32x8;
typedef __attribute__((ext_vector_type(2)))  float  f32x2;

union Frag { bf16x16 v; uint4 q[2]; };

// ---------- problem dims ----------
constexpr int kB = 128, kT = 64, kD = 1024, kM = 2048;
constexpr int kN = kB * kT;            // 8192 tokens

// ---------- GEMM tiling ----------
constexpr int TOK_TILE   = 32;         // tokens per workgroup (A panel resident in LDS)
constexpr int CODE_CHUNK = 64;         // codes per chunk (4 waves x 16)
constexpr int KSTEP      = 32;         // K per WMMA
constexpr int A_STRIDE   = kD + 8;     // 1032 bf16: pad to avoid LDS bank conflicts
constexpr int B_STRIDE   = KSTEP + 8;  // 40 bf16
constexpr int SMEM_BYTES =
    2 * TOK_TILE * A_STRIDE * 2 +      // A hi+lo panels         (132,096 B)
    2 * 2 * 64 * B_STRIDE * 2 +        // B double buffer hi+lo  ( 20,480 B)
    TOK_TILE * 4 * 4 +                 // per-(token,codeSub) sums ( 512 B)
    TOK_TILE * 4 * 8;                  // per-(token,codeSub) keys (1024 B)
static_assert(SMEM_BYTES <= 160 * 1024, "LDS budget (CU mode: 160KB)");

// d_out layout (reference tuple, flattened)
constexpr size_t OFF_VFULL = 0;
constexpr size_t OFF_AFULL = (size_t)kN * 2 * kD;          // 16,777,216
constexpr size_t OFF_VQV   = OFF_AFULL * 2;                // 33,554,432
constexpr size_t OFF_VQA   = OFF_VQV + (size_t)kN * kD;    // 41,943,040
constexpr size_t OFF_LCM   = OFF_VQA + (size_t)kN * kD;    // 50,331,648
constexpr size_t OFF_EQ    = OFF_LCM + 1;                  // 50,331,649

// ---------- CDNA5 async memory->LDS copy (ASYNCcnt-tracked, no VGPR data) ----
// Flat LDS addresses carry the LDS byte offset in addr[31:0] (ISA 10.2), so a
// truncated generic pointer is the VDST address the instruction expects.
__device__ __forceinline__ uint32_t lds_off(const void* p) {
  return (uint32_t)(uintptr_t)p;
}
__device__ __forceinline__ void async_b128(uint32_t lds_byte_addr, const void* gaddr) {
  asm volatile("global_load_async_to_lds_b128 %0, %1, off"
               :: "v"(lds_byte_addr), "v"(gaddr)
               : "memory");
}
__device__ __forceinline__ void wait_async0() {
  asm volatile("s_wait_asynccnt 0" ::: "memory");
}

// =====================================================================
// prep: split embedding into bf16 hi/lo halves + per-code squared norms
// =====================================================================
__global__ void prep_emb_kernel(const float* __restrict__ emb,
                                __bf16* __restrict__ Evh, __bf16* __restrict__ Evl,
                                __bf16* __restrict__ Eah, __bf16* __restrict__ Eal,
                                float* __restrict__ e2v, float* __restrict__ e2a) {
  __shared__ float rv[256], ra[256];
  const int m = blockIdx.x, tid = threadIdx.x;
  float sv = 0.f, sa = 0.f;
  for (int d = tid; d < kD; d += 256) {
    float v = emb[(size_t)m * (2 * kD) + d];          // v_emb = embedding[:, :D]
    __bf16 h = (__bf16)v;
    Evh[(size_t)m * kD + d] = h;
    Evl[(size_t)m * kD + d] = (__bf16)(v - (float)h);
    sv += v * v;
    float a = emb[(size_t)m * (2 * kD) + kD + d];     // a_emb = embedding[:, D:]
    __bf16 h2 = (__bf16)a;
    Eah[(size_t)m * kD + d] = h2;
    Eal[(size_t)m * kD + d] = (__bf16)(a - (float)h2);
    sa += a * a;
  }
  rv[tid] = sv; ra[tid] = sa; __syncthreads();
  for (int s = 128; s > 0; s >>= 1) {
    if (tid < s) { rv[tid] += rv[tid + s]; ra[tid] += ra[tid + s]; }
    __syncthreads();
  }
  if (tid == 0) { e2v[m] = rv[0]; e2a[m] = ra[0]; }
}

// =====================================================================
// prep: split tokens into bf16 hi/lo + per-token squared norms
// blockIdx.y: 0 = video, 1 = audio
// =====================================================================
__global__ void prep_tok_kernel(const float* __restrict__ audio,
                                const float* __restrict__ video,
                                __bf16* __restrict__ Xvh, __bf16* __restrict__ Xvl,
                                __bf16* __restrict__ Xah, __bf16* __restrict__ Xal,
                                float* __restrict__ x2v, float* __restrict__ x2a) {
  __shared__ float red[256];
  const int t = blockIdx.x, tid = threadIdx.x;
  const float* src; __bf16 *xh, *xl; float* x2;
  if (blockIdx.y == 0) { src = video; xh = Xvh; xl = Xvl; x2 = x2v; }
  else                 { src = audio; xh = Xah; xl = Xal; x2 = x2a; }
  float ss = 0.f;
  for (int d = tid; d < kD; d += 256) {
    float v = src[(size_t)t * kD + d];
    __bf16 h = (__bf16)v;
    xh[(size_t)t * kD + d] = h;
    xl[(size_t)t * kD + d] = (__bf16)(v - (float)h);
    ss += v * v;
  }
  red[tid] = ss; __syncthreads();
  for (int s = 128; s > 0; s >>= 1) {
    if (tid < s) red[tid] += red[tid + s];
    __syncthreads();
  }
  if (tid == 0) x2[t] = red[0];
}

// =====================================================================
// Core: fused distance GEMM (split-bf16 fp32 emulation, 3 WMMAs/K-step)
//   S[t][m] = -sqrt(max(x2[t] + e2[m] - 2*dot, 0))
// plus per-token sum(exp(S)) and argmax(S) (== argmin dist).
// All global->LDS staging via GLOBAL_LOAD_ASYNC_TO_LDS_B128 (ASYNCcnt).
// Grid: kN/TOK_TILE blocks x 256 threads (8 waves: 2 token-sub x 4 code-sub)
// =====================================================================
__global__ __launch_bounds__(256, 1)
void dist_gemm_kernel(const __bf16* __restrict__ Xhi, const __bf16* __restrict__ Xlo,
                      const __bf16* __restrict__ Ehi, const __bf16* __restrict__ Elo,
                      const float* __restrict__ x2, const float* __restrict__ e2,
                      float* __restrict__ Sbuf, float* __restrict__ rowsum,
                      int* __restrict__ idxOut) {
  extern __shared__ unsigned char smem_raw[];
  __bf16* Ah = (__bf16*)smem_raw;                       // [TOK_TILE][A_STRIDE]
  __bf16* Al = Ah + TOK_TILE * A_STRIDE;
  __bf16* Bt = Al + TOK_TILE * A_STRIDE;                // [buf][hl][64*B_STRIDE]
  float*  s_sum = (float*)(Bt + 2 * 2 * 64 * B_STRIDE); // [TOK_TILE][4]
  unsigned long long* s_key = (unsigned long long*)(s_sum + TOK_TILE * 4);

  const int tid   = threadIdx.x;
  const int wave  = tid >> 5;
  const int lane  = tid & 31;
  const int tBase = blockIdx.x * TOK_TILE;

  // LDS byte addresses for async copies
  const uint32_t AhB = lds_off(Ah);
  const uint32_t AlB = lds_off(Al);
  const uint32_t BtB = lds_off(Bt);

  if (tid < TOK_TILE * 4) { s_sum[tid] = 0.f; s_key[tid] = 0ull; }

  // ---- Stage A panel (hi+lo): 32 rows x 1024 bf16, async 16B chunks ----
  // 4096 chunks per panel, 16 per thread; 32 async ops/thread (< ASYNCcnt cap)
  #pragma unroll
  for (int j = 0; j < 16; ++j) {
    const int i = j * 256 + tid;
    const int r = i >> 7, c = (i & 127) * 8;
    const size_t g = (size_t)(tBase + r) * kD + c;
    const uint32_t l = (uint32_t)(r * A_STRIDE + c) * 2u;
    async_b128(AhB + l, Xhi + g);
    async_b128(AlB + l, Xlo + g);
  }

  const int tokSub  = wave >> 2;     // 0..1
  const int codeSub = wave & 3;      // 0..3
  const int sub16   = lane & 15;
  const int khalf   = lane >> 4;
  const int aRowOff = (tokSub * 16 + sub16) * A_STRIDE;
  const int bRowOff = (codeSub * 16 + sub16) * B_STRIDE;

  // B tile: 64 code rows x 32 K x (hi,lo) = 512 x 16B chunks; 2 per thread
  const int br = (tid >> 2) & 63;        // code row within chunk
  const int bc = (tid & 3) * 8;          // 8-bf16 column group
  auto loadB_async = [&](int buf, int cb, int ks) {
    const size_t g = (size_t)(cb * CODE_CHUNK + br) * kD + ks * KSTEP + bc;
    const uint32_t lh = BtB + (uint32_t)(((buf * 2 + 0) * 64 + br) * B_STRIDE + bc) * 2u;
    const uint32_t ll = BtB + (uint32_t)(((buf * 2 + 1) * 64 + br) * B_STRIDE + bc) * 2u;
    async_b128(lh, Ehi + g);
    async_b128(ll, Elo + g);
  };

  for (int cb = 0; cb < kM / CODE_CHUNK; ++cb) {
    f32x8 acc;
    #pragma unroll
    for (int r = 0; r < 8; ++r) acc[r] = 0.f;

    loadB_async(0, cb, 0);       // covers A panel on first chunk too
    wait_async0();
    __syncthreads();

    for (int ks = 0; ks < kD / KSTEP; ++ks) {
      const int cur = ks & 1;
      // prefetch next tile: issued after the barrier that retired all reads
      // of buf cur^1, so the in-place overwrite is race-free
      if (ks + 1 < kD / KSTEP) loadB_async(cur ^ 1, cb, ks + 1);

      // A fragment (16x32 bf16): lanes<16 hold K 0-7,16-23; lanes>=16 hold 8-15,24-31
      Frag ah, al, bh, bl;
      const int ae = aRowOff + ks * KSTEP + khalf * 8;
      ah.q[0] = *(const uint4*)&Ah[ae];
      ah.q[1] = *(const uint4*)&Ah[ae + 16];
      al.q[0] = *(const uint4*)&Al[ae];
      al.q[1] = *(const uint4*)&Al[ae + 16];
      const int be  = (cur * 2) * 64 * B_STRIDE + bRowOff + khalf * 8;
      const int beL = be + 64 * B_STRIDE;
      bh.q[0] = *(const uint4*)&Bt[be];
      bh.q[1] = *(const uint4*)&Bt[be + 16];
      bl.q[0] = *(const uint4*)&Bt[beL];
      bl.q[1] = *(const uint4*)&Bt[beL + 16];

      // fp32-emulated dot: hi*hi + hi*lo + lo*hi, one f32 accumulator
      acc = __builtin_amdgcn_wmma_f32_16x16x32_bf16(false, ah.v, false, bh.v, (short)0, acc, false, false);
      acc = __builtin_amdgcn_wmma_f32_16x16x32_bf16(false, ah.v, false, bl.v, (short)0, acc, false, false);
      acc = __builtin_amdgcn_wmma_f32_16x16x32_bf16(false, al.v, false, bh.v, (short)0, acc, false, false);

      wait_async0();       // prefetch landed (latency hidden under WMMAs)
      __syncthreads();     // ... and visible to all waves
    }

    // Epilogue: C layout: element (vgpr r, lane l) -> row r + 8*(l>>4), col l&15
    const int m   = cb * CODE_CHUNK + codeSub * 16 + sub16;
    const float e2m = e2[m];
    #pragma unroll
    for (int r = 0; r < 8; ++r) {
      const int tokLoc = tokSub * 16 + r + 8 * khalf;
      const int t = tBase + tokLoc;
      const float dist = x2[t] + e2m - 2.0f * acc[r];
      const float s = -sqrtf(fmaxf(dist, 0.0f));
      Sbuf[(size_t)t * kM + m] = s;

      float sum = __expf(s);  // s in ~[-36,-28]: exp is ~1e-16, safe in f32
      unsigned int fb = __float_as_uint(s);
      fb ^= (fb & 0x80000000u) ? 0xFFFFFFFFu : 0x80000000u;  // order-preserving map
      unsigned long long key =
          ((unsigned long long)fb << 32) | (unsigned int)(kM - 1 - m);  // ties -> lowest m

      #pragma unroll
      for (int off = 1; off < 16; off <<= 1) {  // reduce 16 cols within each half
        sum += __shfl_xor(sum, off, 32);
        const unsigned int ohi = __shfl_xor((unsigned int)(key >> 32), off, 32);
        const unsigned int olo = __shfl_xor((unsigned int)(key & 0xFFFFFFFFu), off, 32);
        const unsigned long long ok = ((unsigned long long)ohi << 32) | olo;
        if (ok > key) key = ok;
      }
      if (sub16 == 0) {  // unique (tokLoc, codeSub) writer -> deterministic
        s_sum[tokLoc * 4 + codeSub] += sum;
        if (key > s_key[tokLoc * 4 + codeSub]) s_key[tokLoc * 4 + codeSub] = key;
      }
    }
  }
  __syncthreads();

  if (tid < TOK_TILE) {
    float s0 = s_sum[tid * 4 + 0] + s_sum[tid * 4 + 1] +
               s_sum[tid * 4 + 2] + s_sum[tid * 4 + 3];
    unsigned long long k0 = s_key[tid * 4 + 0];
    for (int j = 1; j < 4; ++j) if (s_key[tid * 4 + j] > k0) k0 = s_key[tid * 4 + j];
    rowsum[tBase + tid] = s0;
    idxOut[tBase + tid] = (kM - 1) - (int)(k0 & 0xFFFFFFFFu);
  }
}

// =====================================================================
// pH[b][m] = mean_t softmax_m(S)[b*T+t][m];  LogP = log(pH + 1e-10)
// =====================================================================
__global__ void ph_kernel(const float* __restrict__ Sbuf, const float* __restrict__ rowsum,
                          float* __restrict__ pH, float* __restrict__ LogP) {
  const int g = blockIdx.x * 256 + threadIdx.x;       // [0, kB*kM)
  const int m = g & (kM - 1);
  const int b = g >> 11;
  float acc = 0.f;
  for (int t = 0; t < kT; ++t) {
    const int tok = b * kT + t;
    acc += __expf(Sbuf[(size_t)tok * kM + m]) * (1.0f / rowsum[tok]);
  }
  const float ph = acc * (1.0f / kT);
  pH[g] = ph;
  LogP[g] = logf(ph + 1e-10f);
}

// =====================================================================
// Scode (128x128) = a_pH @ LogV^T + v_pH @ LogA^T via f32 WMMA 16x16x4
// =====================================================================
__global__ __launch_bounds__(256, 1)
void scode_kernel(const float* __restrict__ aPH, const float* __restrict__ vPH,
                  const float* __restrict__ LogV, const float* __restrict__ LogA,
                  float* __restrict__ Scode) {
  const int wave = threadIdx.x >> 5, lane = threadIdx.x & 31;
  const int sub16 = lane & 15, khalf = lane >> 4;
  const int i0 = blockIdx.x * 16;  // 8 blocks cover 128 rows
  const int j0 = wave * 16;        // 8 waves cover 128 cols
#if __has_builtin(__builtin_amdgcn_wmma_f32_16x16x4_f32)
  f32x8 acc;
  #pragma unroll
  for (int r = 0; r < 8; ++r) acc[r] = 0.f;
  const size_t rA = (size_t)(i0 + sub16) * kM;
  const size_t rB = (size_t)(j0 + sub16) * kM;
  for (int k = 0; k < kM; k += 4) {
    const int kk = k + 2 * khalf;  // A 16x4 f32: lane holds K = 2*(l>>4)+{0,1}
    f32x2 a, b;
    a[0] = aPH[rA + kk];  a[1] = aPH[rA + kk + 1];
    b[0] = LogV[rB + kk]; b[1] = LogV[rB + kk + 1];
    acc = __builtin_amdgcn_wmma_f32_16x16x4_f32(false, a, false, b, (short)0, acc, false, false);
    a[0] = vPH[rA + kk];  a[1] = vPH[rA + kk + 1];
    b[0] = LogA[rB + kk]; b[1] = LogA[rB + kk + 1];
    acc = __builtin_amdgcn_wmma_f32_16x16x4_f32(false, a, false, b, (short)0, acc, false, false);
  }
  #pragma unroll
  for (int r = 0; r < 8; ++r)
    Scode[(size_t)(i0 + r + 8 * khalf) * kB + j0 + sub16] = acc[r];
#else
  float accr[8];
  #pragma unroll
  for (int r = 0; r < 8; ++r) accr[r] = 0.f;
  const int j = j0 + sub16;
  for (int k = 0; k < kM; ++k) {
    const float lv = LogV[(size_t)j * kM + k], la = LogA[(size_t)j * kM + k];
    #pragma unroll
    for (int r = 0; r < 8; ++r) {
      const int i = i0 + r + 8 * khalf;
      accr[r] += aPH[(size_t)i * kM + k] * lv + vPH[(size_t)i * kM + k] * la;
    }
  }
  #pragma unroll
  for (int r = 0; r < 8; ++r)
    Scode[(size_t)(i0 + r + 8 * khalf) * kB + j] = accr[r];
#endif
}

// =====================================================================
// Lcmcm = -mean(log(diag(exp(S+Max)) / (rowsum + 1e-5))),  Max = max(-S)
// =====================================================================
__global__ void lcmcm_kernel(const float* __restrict__ Sc, float* __restrict__ out) {
  __shared__ float red[128];
  const int i = threadIdx.x;
  float lmax = -3.4e38f;
  for (int j = 0; j < kB; ++j) lmax = fmaxf(lmax, -Sc[i * kB + j]);
  red[i] = lmax; __syncthreads();
  for (int s = 64; s > 0; s >>= 1) {
    if (i < s) red[i] = fmaxf(red[i], red[i + s]);
    __syncthreads();
  }
  const float Mx = red[0]; __syncthreads();
  float rs = 0.f, dg = 0.f;
  for (int j = 0; j < kB; ++j) {
    const float e = __expf(Sc[i * kB + j] + Mx);
    rs += e;
    if (j == i) dg = e;
  }
  red[i] = logf(dg / (rs + 1e-5f)); __syncthreads();
  for (int s = 64; s > 0; s >>= 1) {
    if (i < s) red[i] += red[i + s];
    __syncthreads();
  }
  if (i == 0) out[0] = -red[0] / (float)kB;
}

// =====================================================================
// Gather outputs: v_full/a_full = embedding[idx]; vq_v/vq_a = halves.
// Also histogram codebook usage per batch (int atomics: deterministic).
// =====================================================================
__global__ void gather_kernel(const float* __restrict__ emb,
                              const int* __restrict__ idxv, const int* __restrict__ idxa,
                              float* __restrict__ v_full, float* __restrict__ a_full,
                              float* __restrict__ vq_v, float* __restrict__ vq_a,
                              int* __restrict__ cv, int* __restrict__ ca) {
  const int t = blockIdx.x, tid = threadIdx.x;
  const int iv = idxv[t], ia = idxa[t];
  const float4* ev = (const float4*)(emb + (size_t)iv * (2 * kD));
  const float4* ea = (const float4*)(emb + (size_t)ia * (2 * kD));
  float4* vf = (float4*)(v_full + (size_t)t * (2 * kD));
  float4* af = (float4*)(a_full + (size_t)t * (2 * kD));
  float4* qv = (float4*)(vq_v + (size_t)t * kD);
  float4* qa = (float4*)(vq_a + (size_t)t * kD);
  for (int c = tid; c < 512; c += 256) {       // 2048 floats = 512 float4
    const float4 x = ev[c];
    vf[c] = x;
    if (c < 256) qv[c] = x;                    // vq_v = embedding[iv][:D]
    const float4 y = ea[c];
    af[c] = y;
    if (c >= 256) qa[c - 256] = y;             // vq_a = embedding[ia][D:]
  }
  if (tid == 0) {
    const int b = t >> 6;                      // t / kT
    atomicAdd(&cv[b * kM + iv], 1);
    atomicAdd(&ca[b * kM + ia], 1);
  }
}

// =====================================================================
// equal_num = sum_b [argmax(counts_v[b]) == argmax(counts_a[b])]
// =====================================================================
__global__ void eqnum_kernel(const int* __restrict__ cv, const int* __restrict__ ca,
                             float* __restrict__ out) {
  __shared__ int red[128];
  const int b = threadIdx.x;
  int mv = -1, iv = 0, ma = -1, ia = 0;
  for (int m = 0; m < kM; ++m) {
    const int c0 = cv[(size_t)b * kM + m];
    if (c0 > mv) { mv = c0; iv = m; }          // strict '>' keeps first max
    const int c1 = ca[(size_t)b * kM + m];
    if (c1 > ma) { ma = c1; ia = m; }
  }
  red[b] = (iv == ia) ? 1 : 0; __syncthreads();
  for (int s = 64; s > 0; s >>= 1) {
    if (b < s) red[b] += red[b + s];
    __syncthreads();
  }
  if (b == 0) out[0] = (float)red[0];
}

// =====================================================================
// Host launcher
// =====================================================================
extern "C" void kernel_launch(void* const* d_in, const int* in_sizes, int n_in,
                              void* d_out, int out_size, void* d_ws, size_t ws_size,
                              hipStream_t stream) {
  (void)in_sizes; (void)n_in; (void)out_size; (void)ws_size;
  const float* audio = (const float*)d_in[0];
  const float* video = (const float*)d_in[1];
  const float* emb   = (const float*)d_in[2];
  float* out = (float*)d_out;

  // ---- workspace carve-out (~158 MB total) ----
  uint8_t* ws = (uint8_t*)d_ws;
  size_t o = 0;
  auto take = [&](size_t bytes) -> void* {
    o = (o + 255) & ~(size_t)255;
    void* p = ws + o;
    o += bytes;
    return p;
  };
  float*  Sbuf = (float*) take((size_t)kN * kM * 4);       // 64 MB (reused v->a)
  __bf16* Evh  = (__bf16*)take((size_t)kM * kD * 2);
  __bf16* Evl  = (__bf16*)take((size_t)kM * kD * 2);
  __bf16* Eah  = (__bf16*)take((size_t)kM * kD * 2);
  __bf16* Eal  = (__bf16*)take((size_t)kM * kD * 2);
  __bf16* Xvh  = (__bf16*)take((size_t)kN * kD * 2);
  __bf16* Xvl  = (__bf16*)take((size_t)kN * kD * 2);
  __bf16* Xah  = (__bf16*)take((size_t)kN * kD * 2);
  __bf16* Xal  = (__bf16*)take((size_t)kN * kD * 2);
  float*  e2v  = (float*) take((size_t)kM * 4);
  float*  e2a  = (float*) take((size_t)kM * 4);
  float*  x2v  = (float*) take((size_t)kN * 4);
  float*  x2a  = (float*) take((size_t)kN * 4);
  float*  rsv  = (float*) take((size_t)kN * 4);
  float*  rsa  = (float*) take((size_t)kN * 4);
  int*    idxv = (int*)   take((size_t)kN * 4);
  int*    idxa = (int*)   take((size_t)kN * 4);
  float*  pHv  = (float*) take((size_t)kB * kM * 4);
  float*  pHa  = (float*) take((size_t)kB * kM * 4);
  float*  LogV = (float*) take((size_t)kB * kM * 4);
  float*  LogA = (float*) take((size_t)kB * kM * 4);
  float*  Scod = (float*) take((size_t)kB * kB * 4);
  int*    cnts = (int*)   take((size_t)2 * kB * kM * 4);   // cv | ca contiguous
  int* cv = cnts;
  int* ca = cnts + (size_t)kB * kM;

  hipFuncSetAttribute(reinterpret_cast<const void*>(dist_gemm_kernel),
                      hipFuncAttributeMaxDynamicSharedMemorySize, SMEM_BYTES);

  // ---- pipeline (all stream-ordered; graph-capture safe) ----
  hipLaunchKernelGGL(prep_emb_kernel, dim3(kM), dim3(256), 0, stream,
                     emb, Evh, Evl, Eah, Eal, e2v, e2a);
  hipLaunchKernelGGL(prep_tok_kernel, dim3(kN, 2), dim3(256), 0, stream,
                     audio, video, Xvh, Xvl, Xah, Xal, x2v, x2a);
  hipMemsetAsync(cnts, 0, (size_t)2 * kB * kM * 4, stream);

  // video: dist + stats, then pH (Sbuf consumed before reuse)
  hipLaunchKernelGGL(dist_gemm_kernel, dim3(kN / TOK_TILE), dim3(256), SMEM_BYTES, stream,
                     Xvh, Xvl, Evh, Evl, x2v, e2v, Sbuf, rsv, idxv);
  hipLaunchKernelGGL(ph_kernel, dim3(kB * kM / 256), dim3(256), 0, stream,
                     Sbuf, rsv, pHv, LogV);
  // audio
  hipLaunchKernelGGL(dist_gemm_kernel, dim3(kN / TOK_TILE), dim3(256), SMEM_BYTES, stream,
                     Xah, Xal, Eah, Eal, x2a, e2a, Sbuf, rsa, idxa);
  hipLaunchKernelGGL(ph_kernel, dim3(kB * kM / 256), dim3(256), 0, stream,
                     Sbuf, rsa, pHa, LogA);

  hipLaunchKernelGGL(scode_kernel, dim3(kB / 16), dim3(256), 0, stream,
                     pHa, pHv, LogV, LogA, Scod);
  hipLaunchKernelGGL(lcmcm_kernel, dim3(1), dim3(128), 0, stream, Scod, out + OFF_LCM);

  hipLaunchKernelGGL(gather_kernel, dim3(kN), dim3(256), 0, stream,
                     emb, idxv, idxa,
                     out + OFF_VFULL, out + OFF_AFULL, out + OFF_VQV, out + OFF_VQA,
                     cv, ca);
  hipLaunchKernelGGL(eqnum_kernel, dim3(1), dim3(128), 0, stream, cv, ca, out + OFF_EQ);
}
